// MiniBindingAttention_37288906064319
// MI455X (gfx1250) — matrix operand order repaired and checked by reading, not compiled, not run on hardware
//
#include <hip/hip_runtime.h>

typedef __attribute__((ext_vector_type(16))) _Float16 v16h;
typedef __attribute__((ext_vector_type(8)))  _Float16 v8h;
typedef __attribute__((ext_vector_type(8)))  float    v8f;

namespace {
constexpr int kB = 2, kT = 2048, kD = 1024, kH = 16, kHD = 64;
constexpr int kWaves = 8;  // waves per block (wave32)
constexpr int kKT = 32;    // keys per iteration
}

// ---------------------------------------------------------------------------
// Pre-pass: x (f32) -> sign-gated f16 buffers.
//   qh[b,t,h,d] = f16(x ^ (signbit(bvq)^signbit(bvk)))   (gating folded into Q)
//   kh[b,t,h,d] = f16(x)
//   vt[b,h,d,t] = f16(x ^ signbit(bvv))                  (transposed for V^T)
// One block handles a 64t x 64d tile of one head; transpose via LDS.
// ---------------------------------------------------------------------------
__global__ __launch_bounds__(256) void mba_convert_kernel(
    const float* __restrict__ x, const float* __restrict__ bvq,
    const float* __restrict__ bvk, const float* __restrict__ bvv,
    _Float16* __restrict__ qh, _Float16* __restrict__ kh,
    _Float16* __restrict__ vt) {
  __shared__ __align__(32) _Float16 vtile[64 * 64];  // [t][d]
  __shared__ unsigned gqk[kHD], gv[kHD];

  const int tid = threadIdx.x;
  const int b = blockIdx.x / (kT / 64);
  const int t0 = (blockIdx.x % (kT / 64)) * 64;
  const int h = blockIdx.y;

  if (tid < kHD) {
    const unsigned uq = __float_as_uint(bvq[h * kHD + tid]);
    const unsigned uk = __float_as_uint(bvk[h * kHD + tid]);
    const unsigned uv = __float_as_uint(bvv[h * kHD + tid]);
    gqk[tid] = (uq ^ uk) & 0x80000000u;
    gv[tid] = uv & 0x80000000u;
  }
  __syncthreads();

  {  // coalesced read of 64x64 f32 tile; 16 floats per thread
    const int r = tid >> 2;            // t row 0..63
    const int c0 = (tid & 3) * 16;     // d col base
    const float* xr = x + ((size_t)(b * kT + t0 + r)) * kD + h * kHD + c0;
    v16h qv, kv;
#pragma unroll
    for (int j = 0; j < 16; ++j) {
      const unsigned u = __float_as_uint(xr[j]);
      const int d = c0 + j;
      kv[j] = (_Float16)__uint_as_float(u);
      qv[j] = (_Float16)__uint_as_float(u ^ gqk[d]);
      vtile[r * 64 + d] = (_Float16)__uint_as_float(u ^ gv[d]);
    }
    const size_t idx = ((size_t)(b * kT + t0 + r)) * kD + h * kHD + c0;
    *(v16h*)&qh[idx] = qv;  // 32B contiguous stores
    *(v16h*)&kh[idx] = kv;
  }
  __syncthreads();

  {  // transposed V write: contiguous 32B along t
    const int d = tid >> 2;
    const int ts = (tid & 3) * 16;
    v16h vv;
#pragma unroll
    for (int j = 0; j < 16; ++j) vv[j] = vtile[(ts + j) * 64 + d];
    *(v16h*)&vt[((size_t)((b * kH + h) * kHD + d)) * kT + t0 + ts] = vv;
  }
}

// ---------------------------------------------------------------------------
// Main kernel (fast path). Per wave: 32 queries. Per iteration: 32 keys,
// K/V tiles double-buffered in LDS via global_load_async_to_lds_b128.
// Computes S^T = K @ Q^T and O^T = V^T @ P^T so that the sigmoid staging
// stores and the final output stores are contiguous b128s.
// ---------------------------------------------------------------------------
__global__ __launch_bounds__(256) void mba_fast_kernel(
    const _Float16* __restrict__ qh, const _Float16* __restrict__ kh,
    const _Float16* __restrict__ vt, float* __restrict__ out) {
  __shared__ __align__(32) _Float16 Kl[2][kKT * kHD];      // [key][dim]
  __shared__ __align__(32) _Float16 Vl[2][kHD * kKT];      // [dim][key]
  __shared__ __align__(32) _Float16 Pl[kWaves][32 * kKT];  // [query][key]

  const int tid = threadIdx.x;
  const int lane = tid & 31;
  const int wave = tid >> 5;
  const int nlo = lane & 15;
  const int hi = lane >> 4;
  const int off = hi * 8;

  const int b = blockIdx.x >> 4;
  const int h = blockIdx.x & 15;
  const int q0 = (blockIdx.y * kWaves + wave) * 32;

  // ---- Q as B-fragments (32x16 B layout: lane=N=query, contiguous dims) ----
  v16h qb[2][2];
#pragma unroll
  for (int t = 0; t < 2; ++t)
#pragma unroll
    for (int c = 0; c < 2; ++c)
      qb[t][c] = *(const v16h*)(qh +
                                ((size_t)(b * kT + q0 + 16 * t + nlo)) * kD +
                                h * kHD + 32 * c + hi * 16);

  v8f o[4][2] = {};  // O^T accumulators: [dim tile][query tile]

  // per-thread async copy assignments (one b128 each for K and V per tile)
  const int kkey = tid >> 3, kdb = (tid & 7) * 8;   // K: [key][dim]
  const int vdd = tid >> 2, vts = (tid & 3) * 8;    // V: [dim][key]
  const _Float16* ksrc = kh + ((size_t)(b * kT)) * kD + h * kHD;
  const _Float16* vsrc = vt + ((size_t)((b * kH + h) * kHD)) * kT;

#define ISSUE_FILL(buf, k0v)                                                   \
  do {                                                                         \
    unsigned kdst = (unsigned)(uintptr_t)&Kl[buf][kkey * kHD + kdb];           \
    unsigned long long kga =                                                   \
        (unsigned long long)(uintptr_t)(ksrc + (size_t)((k0v) + kkey) * kD +   \
                                        kdb);                                  \
    asm volatile("global_load_async_to_lds_b128 %0, %1, off" ::"v"(kdst),      \
                 "v"(kga)                                                      \
                 : "memory");                                                  \
    unsigned vdst = (unsigned)(uintptr_t)&Vl[buf][vdd * kKT + vts];            \
    unsigned long long vga =                                                   \
        (unsigned long long)(uintptr_t)(vsrc + (size_t)vdd * kT + (k0v) +      \
                                        vts);                                  \
    asm volatile("global_load_async_to_lds_b128 %0, %1, off" ::"v"(vdst),      \
                 "v"(vga)                                                      \
                 : "memory");                                                  \
  } while (0)

  const int nkt = (blockIdx.y + 1) * 8;  // key blocks covering this q range
  ISSUE_FILL(0, 0);

  for (int kt = 0; kt < nkt; ++kt) {
    const int k0 = kt * kKT;
    const int cur = kt & 1;
    asm volatile("s_wait_asynccnt 0" ::: "memory");  // own copies landed
    __syncthreads();                                 // everyone's copies landed
    if (kt + 1 < nkt) ISSUE_FILL(1 - cur, k0 + kKT); // prefetch next tile

    if (k0 > q0 + 31) continue;  // wave-uniform causal early-out

    const _Float16* Kb = &Kl[cur][0];
    const _Float16* Vb = &Vl[cur][0];

    // ---- S^T = K @ Q^T : 4 tiles (2 key subtiles x 2 query subtiles) ----
#pragma unroll
    for (int nt = 0; nt < 2; ++nt) {
      v16h ka[2];  // A layout: lane -> key row, contiguous 16 dims per half
#pragma unroll
      for (int c = 0; c < 2; ++c) {
        const v8h a0 = *(const v8h*)&Kb[(nt * 16 + nlo) * kHD + 32 * c + hi * 16];
        const v8h a1 =
            *(const v8h*)&Kb[(nt * 16 + nlo) * kHD + 32 * c + hi * 16 + 8];
        ka[c] = __builtin_shufflevector(a0, a1, 0, 1, 2, 3, 4, 5, 6, 7, 8, 9,
                                        10, 11, 12, 13, 14, 15);
      }
#pragma unroll
      for (int t = 0; t < 2; ++t) {
        v8f s = {};
        s = __builtin_amdgcn_wmma_f32_16x16x32_f16(false, ka[0], false,
                                                   qb[t][0], (short)0, s,
                                                   false, false);
        s = __builtin_amdgcn_wmma_f32_16x16x32_f16(false, ka[1], false,
                                                   qb[t][1], (short)0, s,
                                                   false, false);
        // S^T C layout: element i -> (key = k0+nt*16+hi*8+i, query = q0+16t+nlo)
        const int qcol = q0 + 16 * t + nlo;
        const int krow0 = k0 + nt * 16 + hi * 8;
        v8h p;
#pragma unroll
        for (int i = 0; i < 8; ++i) {
          const float a = (krow0 + i <= qcol)
                              ? (1.0f / (1.0f + __expf(-0.5f * s[i])))
                              : 0.0f;  // 4*HD^-0.5 = 0.5
          p[i] = (_Float16)a;
        }
        // contiguous 16B store: row=query, 8 consecutive keys
        *(v8h*)&Pl[wave][(16 * t + nlo) * kKT + nt * 16 + hi * 8] = p;
      }
    }

    // ---- O^T += V^T @ P^T ----
    v16h pb[2];  // B layout: lane=N=query, contiguous 16 keys per half
#pragma unroll
    for (int t = 0; t < 2; ++t)
      pb[t] = *(const v16h*)&Pl[wave][(16 * t + nlo) * kKT + hi * 16];
#pragma unroll
    for (int nd = 0; nd < 4; ++nd) {
      const v8h a0 = *(const v8h*)&Vb[(nd * 16 + nlo) * kKT + hi * 16];
      const v8h a1 = *(const v8h*)&Vb[(nd * 16 + nlo) * kKT + hi * 16 + 8];
      const v16h va = __builtin_shufflevector(a0, a1, 0, 1, 2, 3, 4, 5, 6, 7,
                                              8, 9, 10, 11, 12, 13, 14, 15);
#pragma unroll
      for (int t = 0; t < 2; ++t)
        o[nd][t] = __builtin_amdgcn_wmma_f32_16x16x32_f16(
            false, va, false, pb[t], (short)0, o[nd][t], false, false);
    }
  }
#undef ISSUE_FILL

  // ---- O^T C layout: element i -> (dim = nd*16+hi*8+i, query = 16t+nlo)
  // 8 consecutive dims per lane => contiguous 32B global stores.
#pragma unroll
  for (int nd = 0; nd < 4; ++nd)
#pragma unroll
    for (int t = 0; t < 2; ++t) {
      float* dst = out + ((size_t)(b * kT + q0 + 16 * t + nlo)) * kD +
                   h * kHD + nd * 16 + hi * 8;
      *(v8f*)dst = o[nd][t];
    }
}

// ---------------------------------------------------------------------------
// Fallback (self-contained) if workspace is too small.
// ---------------------------------------------------------------------------
__global__ __launch_bounds__(256) void mba_fallback_kernel(
    const float* __restrict__ x, const float* __restrict__ bvq,
    const float* __restrict__ bvk, const float* __restrict__ bvv,
    float* __restrict__ out) {
  __shared__ __align__(32) _Float16 Klds[kKT * kHD];
  __shared__ __align__(32) _Float16 Vlds[kHD * kKT];
  __shared__ __align__(32) _Float16 Plds[kWaves][16 * kKT];
  __shared__ unsigned gmask[kHD];
  __shared__ unsigned vmask[kHD];

  const int tid = threadIdx.x;
  const int lane = tid & 31;
  const int wave = tid >> 5;
  const int nlo = lane & 15;
  const int hi = lane >> 4;
  const int b = blockIdx.x >> 4;
  const int h = blockIdx.x & 15;
  const int q0 = (blockIdx.y * kWaves + wave) * 16;

  if (tid < kHD) {
    const unsigned uq = __float_as_uint(bvq[h * kHD + tid]);
    const unsigned uk = __float_as_uint(bvk[h * kHD + tid]);
    const unsigned uv = __float_as_uint(bvv[h * kHD + tid]);
    gmask[tid] = (uq ^ uk) & 0x80000000u;
    vmask[tid] = uv & 0x80000000u;
  }
  __syncthreads();

  v16h qa[2];
  {
    const float* xq = x + ((size_t)(b * kT + q0 + nlo)) * kD + h * kHD;
    const int off = hi * 8;
#pragma unroll
    for (int c = 0; c < 2; ++c)
#pragma unroll
      for (int g = 0; g < 2; ++g) {
        const int dbase = 32 * c + 16 * hi + 8 * g;
#pragma unroll
        for (int j = 0; j < 8; ++j) {
          const int d = dbase + j;
          qa[c][g * 8 + j] = (_Float16)__uint_as_float(
              __float_as_uint(xq[d]) ^ gmask[d]);
        }
      }
  }

  v8f o[4] = {};
  const int nkt = blockIdx.y * 2 + 2;
  for (int kt = 0; kt < nkt; ++kt) {
    const int k0 = kt * kKT;
    __syncthreads();
    {
      const int kk = tid >> 3;
      const int dbase = (tid & 7) * 8;
      const float* xk =
          x + ((size_t)(b * kT + k0 + kk)) * kD + h * kHD + dbase;
#pragma unroll
      for (int j = 0; j < 8; ++j) {
        const float xv = xk[j];
        const int d = dbase + j;
        Klds[kk * kHD + d] = (_Float16)xv;
        Vlds[d * kKT + kk] =
            (_Float16)__uint_as_float(__float_as_uint(xv) ^ vmask[d]);
      }
    }
    __syncthreads();
    if (k0 > q0 + 15) continue;

#pragma unroll
    for (int nt = 0; nt < 2; ++nt) {
      v8f s = {};
#pragma unroll
      for (int c = 0; c < 2; ++c) {
        const v16h kb =
            *(const v16h*)&Klds[(nt * 16 + nlo) * kHD + 32 * c + hi * 16];
        s = __builtin_amdgcn_wmma_f32_16x16x32_f16(false, qa[c], false, kb,
                                                   (short)0, s, false, false);
      }
      const int kcol = k0 + nt * 16 + nlo;
#pragma unroll
      for (int i = 0; i < 8; ++i) {
        const int qrow = q0 + hi * 8 + i;
        const float a = (kcol <= qrow)
                            ? (1.0f / (1.0f + __expf(-0.5f * s[i])))
                            : 0.0f;
        Plds[wave][(hi * 8 + i) * kKT + nt * 16 + nlo] = (_Float16)a;
      }
    }
    const int off = hi * 8;
    const v8h p0 = *(const v8h*)&Plds[wave][nlo * kKT + hi * 16];
    const v8h p1 = *(const v8h*)&Plds[wave][nlo * kKT + hi * 16 + 8];
    const v16h pa = __builtin_shufflevector(p0, p1, 0, 1, 2, 3, 4, 5, 6, 7, 8,
                                            9, 10, 11, 12, 13, 14, 15);
#pragma unroll
    for (int nd = 0; nd < 4; ++nd) {
      const v16h vb = *(const v16h*)&Vlds[(nd * 16 + nlo) * kKT + hi * 16];
      o[nd] = __builtin_amdgcn_wmma_f32_16x16x32_f16(false, pa, false, vb,
                                                     (short)0, o[nd], false,
                                                     false);
    }
  }
  float* op = out + ((size_t)(b * kT + q0 + hi * 8)) * kD + h * kHD + nlo;
#pragma unroll
  for (int nd = 0; nd < 4; ++nd)
#pragma unroll
    for (int i = 0; i < 8; ++i) op[(size_t)i * kD + nd * 16] = o[nd][i];
}

extern "C" void kernel_launch(void* const* d_in, const int* in_sizes, int n_in,
                              void* d_out, int out_size, void* d_ws,
                              size_t ws_size, hipStream_t stream) {
  const float* x = (const float*)d_in[0];
  const float* bq = (const float*)d_in[1];
  const float* bk = (const float*)d_in[2];
  const float* bv = (const float*)d_in[3];
  float* out = (float*)d_out;

  const size_t elems = (size_t)kB * kT * kD;  // 4M elements
  const size_t need = 3 * elems * 2;          // qh + kh + vt, f16

  if (ws_size >= need) {
    _Float16* qh = (_Float16*)d_ws;
    _Float16* kh = qh + elems;
    _Float16* vt = kh + elems;
    mba_convert_kernel<<<dim3(kB * (kT / 64), kH), 256, 0, stream>>>(
        x, bq, bk, bv, qh, kh, vt);
    mba_fast_kernel<<<dim3(kB * kH, kT / (kWaves * 32)), 256, 0, stream>>>(
        qh, kh, vt, out);
  } else {
    mba_fallback_kernel<<<dim3(kB * kH, kT / (kWaves * 16)), 256, 0, stream>>>(
        x, bq, bk, bv, out);
  }
}
